// GCN_encoder_spmm_41918880809102
// MI455X (gfx1250) — matrix operand order repaired and checked by reading, not compiled, not run on hardware
//
#include <hip/hip_runtime.h>

typedef __attribute__((ext_vector_type(16))) __bf16 v16bf;
typedef __attribute__((ext_vector_type(8)))  __bf16 v8bf;
typedef __attribute__((ext_vector_type(8)))  float  v8f;

#define IN_DIM  256
#define OUT_DIM 128

// Split W (fp32) into bf16 hi/lo planes once per launch (32K elements, trivial).
__global__ void __launch_bounds__(256) prep_w_kernel(const float* __restrict__ W,
                                                     __bf16* __restrict__ Whi,
                                                     __bf16* __restrict__ Wlo,
                                                     int n) {
    int i = blockIdx.x * 256 + threadIdx.x;
    if (i < n) {
        float v  = W[i];
        __bf16 h = (__bf16)v;
        Whi[i] = h;
        Wlo[i] = (__bf16)(v - (float)h);
    }
}

__global__ void __launch_bounds__(256) zero_kernel(float4* __restrict__ p, int n4) {
    int i = blockIdx.x * 256 + threadIdx.x;
    int stride = gridDim.x * 256;
    float4 z = make_float4(0.f, 0.f, 0.f, 0.f);
    for (; i < n4; i += stride) p[i] = z;
}

// h = x @ W^T via bf16x3 split WMMA (near-fp32 accuracy, bf16 matrix-pipe rate).
// One block = 16 rows of x; 8 waves cover OUT_DIM=128 as 8 N-tiles of 16.
__global__ void __launch_bounds__(256) gemm_kernel(const float*  __restrict__ x,
                                                   const __bf16* __restrict__ Whi,
                                                   const __bf16* __restrict__ Wlo,
                                                   float* __restrict__ h,
                                                   int n_nodes) {
    __shared__ __attribute__((aligned(16))) __bf16 xh[16 * IN_DIM];
    __shared__ __attribute__((aligned(16))) __bf16 xl[16 * IN_DIM];

    const int m0  = blockIdx.x * 16;
    const int tid = threadIdx.x;

    // Stage 16x256 fp32 tile of x -> bf16 hi/lo in LDS (coalesced b32 reads).
    for (int i = tid; i < 16 * IN_DIM; i += 256) {
        int r = i >> 8;          // IN_DIM == 256
        int c = i & (IN_DIM - 1);
        float v = 0.f;
        if (m0 + r < n_nodes) v = x[(size_t)(m0 + r) * IN_DIM + c];
        __bf16 hi = (__bf16)v;
        xh[i] = hi;
        xl[i] = (__bf16)(v - (float)hi);
    }
    __syncthreads();

    const int wave = tid >> 5;
    const int lane = tid & 31;
    const int n0   = wave * 16;              // 8 waves x 16 cols = 128 = OUT_DIM
    const int mr   = lane & 15;
    const int khA  = (lane >> 4) * 8;        // A: lanes 0-15 -> K 0-7/16-23; 16-31 -> 8-15/24-31
    const int khB  = (lane >> 4) * 8;        // B: lanes 0-15 -> K 0-7/16-23; 16-31 -> 8-15/24-31

    const __bf16* pxh = xh + mr * IN_DIM + khA;
    const __bf16* pxl = xl + mr * IN_DIM + khA;
    const __bf16* pwh = Whi + (size_t)(n0 + mr) * IN_DIM + khB;
    const __bf16* pwl = Wlo + (size_t)(n0 + mr) * IN_DIM + khB;

    v8f acc = {};
#pragma unroll
    for (int kc = 0; kc < IN_DIM; kc += 32) {
        v8bf ah0 = *(const v8bf*)(pxh + kc);
        v8bf ah1 = *(const v8bf*)(pxh + kc + 16);
        v8bf al0 = *(const v8bf*)(pxl + kc);
        v8bf al1 = *(const v8bf*)(pxl + kc + 16);
        v8bf bh0 = *(const v8bf*)(pwh + kc);
        v8bf bh1 = *(const v8bf*)(pwh + kc + 16);
        v8bf bl0 = *(const v8bf*)(pwl + kc);
        v8bf bl1 = *(const v8bf*)(pwl + kc + 16);

        v16bf ah = __builtin_shufflevector(ah0, ah1, 0,1,2,3,4,5,6,7,8,9,10,11,12,13,14,15);
        v16bf al = __builtin_shufflevector(al0, al1, 0,1,2,3,4,5,6,7,8,9,10,11,12,13,14,15);
        v16bf bh = __builtin_shufflevector(bh0, bh1, 0,1,2,3,4,5,6,7,8,9,10,11,12,13,14,15);
        v16bf bl = __builtin_shufflevector(bl0, bl1, 0,1,2,3,4,5,6,7,8,9,10,11,12,13,14,15);

        // (xh+xl)(wh+wl) ~= xh*wh + xh*wl + xl*wh  (drop xl*wl ~ 2^-16 rel)
        acc = __builtin_amdgcn_wmma_f32_16x16x32_bf16(false, ah, false, bh, (short)0, acc, false, false);
        acc = __builtin_amdgcn_wmma_f32_16x16x32_bf16(false, ah, false, bl, (short)0, acc, false, false);
        acc = __builtin_amdgcn_wmma_f32_16x16x32_bf16(false, al, false, bh, (short)0, acc, false, false);
    }

    // C/D layout: VGPR v -> row m0+v (lanes 0-15) / m0+v+8 (lanes 16-31), col n0+lane%16.
    const int colh  = n0 + mr;
    const int rbase = m0 + (lane >> 4) * 8;
#pragma unroll
    for (int v = 0; v < 8; ++v) {
        int r = rbase + v;
        if (r < n_nodes) h[(size_t)r * OUT_DIM + colh] = acc[v];
    }
}

// SpMM: out[row[e]] += val[e] * h[col[e]]; one wave per edge, 4 floats/lane.
__global__ void __launch_bounds__(256) spmm_kernel(const int*   __restrict__ erow,
                                                   const int*   __restrict__ ecol,
                                                   const float* __restrict__ eval,
                                                   const float* __restrict__ h,
                                                   float* __restrict__ out,
                                                   int n_edges) {
    const int lane   = threadIdx.x & 31;
    const int wave   = (blockIdx.x * blockDim.x + threadIdx.x) >> 5;
    const int nwaves = (gridDim.x * blockDim.x) >> 5;

    for (int e = wave; e < n_edges; e += nwaves) {
        int   r = erow[e];
        int   c = ecol[e];
        float v = eval[e];
        const float4 hv = *(const float4*)(h + (size_t)c * OUT_DIM + lane * 4);
        float* po = out + (size_t)r * OUT_DIM + lane * 4;
        unsafeAtomicAdd(po + 0, hv.x * v);
        unsafeAtomicAdd(po + 1, hv.y * v);
        unsafeAtomicAdd(po + 2, hv.z * v);
        unsafeAtomicAdd(po + 3, hv.w * v);
    }
}

extern "C" void kernel_launch(void* const* d_in, const int* in_sizes, int n_in,
                              void* d_out, int out_size, void* d_ws, size_t ws_size,
                              hipStream_t stream) {
    const float* x    = (const float*)d_in[0];
    const float* W    = (const float*)d_in[1];
    const int*   erow = (const int*)d_in[2];
    const int*   ecol = (const int*)d_in[3];
    const float* eval = (const float*)d_in[4];
    float*       out  = (float*)d_out;

    const int n_nodes = in_sizes[0] / IN_DIM;   // 100000
    const int wn      = in_sizes[1];            // 128*256
    const int n_edges = in_sizes[2];            // 1600000

    // Workspace layout: h[f32 n_nodes*128] | Whi[bf16 wn] | Wlo[bf16 wn]
    char*   ws      = (char*)d_ws;
    float*  h       = (float*)ws;
    size_t  h_bytes = ((size_t)n_nodes * OUT_DIM * sizeof(float) + 255) & ~(size_t)255;
    size_t  w_bytes = ((size_t)wn * sizeof(__bf16) + 255) & ~(size_t)255;
    __bf16* Whi     = (__bf16*)(ws + h_bytes);
    __bf16* Wlo     = (__bf16*)(ws + h_bytes + w_bytes);

    prep_w_kernel<<<(wn + 255) / 256, 256, 0, stream>>>(W, Whi, Wlo, wn);

    int mblocks = (n_nodes + 15) / 16;
    gemm_kernel<<<mblocks, 256, 0, stream>>>(x, Whi, Wlo, h, n_nodes);

    zero_kernel<<<1024, 256, 0, stream>>>((float4*)out, out_size / 4);

    spmm_kernel<<<4096, 256, 0, stream>>>(erow, ecol, eval, h, out, n_edges);
}
